// SpiralEncoder_2808908612154
// MI455X (gfx1250) — matrix operand order repaired and checked by reading, not compiled, not run on hardware
//
#include <hip/hip_runtime.h>
#include <hip/hip_bf16.h>

// ---------------------------------------------------------------------------
// CDNA5 / gfx1250 SpiralEncoder
//   * gather + GEMM + ELU fused per block; GEMM via v_wmma_f32_16x16x32_bf16
//   * W staged to LDS by the Tensor Data Mover (tensor_load_to_lds),
//     double-buffered, synced with s_wait_tensorcnt + workgroup barrier
//   * one wave owns a 16 x C_out strip: A fragment gathered once, reused
//     across all N tiles (register-resident accumulators)
//   * sparse pooling as a light elementwise kernel
//   * latent projection: split-K WMMA + global_atomic_add_f32 accumulation
// ---------------------------------------------------------------------------

typedef __attribute__((ext_vector_type(16))) __bf16        v16bf;
typedef __attribute__((ext_vector_type(8)))  float         v8f;
typedef __attribute__((ext_vector_type(4)))  unsigned int  u32x4;
typedef __attribute__((ext_vector_type(8)))  unsigned int  u32x8;

__device__ __forceinline__ __bf16 f2bf(float f) {
    // round-to-nearest-even fp32 -> bf16
    unsigned int u = __builtin_bit_cast(unsigned int, f);
    unsigned int r = (u + 0x7FFFu + ((u >> 16) & 1u)) >> 16;
    unsigned short s = (unsigned short)r;
    return __builtin_bit_cast(__bf16, s);
}

// ---------------------------------------------------------------------------
// SpiralConv block: h[b,v,co] = elu( sum_k gather(x)[b,v,k] * W[k,co] + b[co] )
//   rows m = b*V + v (M = 64*V, multiple of 64 => full 4-wave workgroups)
//   k = s*CIN + c ; gathered value = x[b, idx[v,s], c]
// One wave computes a 16 x COUT strip; K loop in steps of 32 (bf16 WMMA).
// W K-chunks staged to LDS by TDM (double buffered).
// ---------------------------------------------------------------------------
template <int CIN, int COUT>
__global__ void spiral_wmma_kernel(const float* __restrict__ x,
                                   const int*   __restrict__ idx,
                                   const float* __restrict__ W,
                                   const float* __restrict__ bias,
                                   float*       __restrict__ h,
                                   int V) {
    constexpr int S    = 9;
    constexpr int KTOT = CIN * S;
    constexpr int NT   = COUT / 16;   // N tiles per strip
    constexpr int KC   = 32;          // K depth per WMMA / per TDM chunk

    __shared__ float wlds[2][KC * COUT];

    const int  lane   = threadIdx.x & 31;
    const int  waveIn = threadIdx.x >> 5;
    const long wave   = (long)blockIdx.x * (blockDim.x >> 5) + waveIn;

    const int laneHi = lane >> 4;     // 0 or 1
    const int nn     = lane & 15;

    // ---- TDM descriptor build + issue: W[kk .. kk+rows) x COUT -> LDS ----
    auto tdm_issue = [&](int kk, int buf) {
        const int rows = ((KTOT - kk) < KC) ? (KTOT - kk) : KC;
        const unsigned lds_off =
            (unsigned)(unsigned long long)(void*)&wlds[buf][0];
        const unsigned long long ga =
            (unsigned long long)(const void*)(W + (long)kk * COUT);
        u32x4 g0;
        g0[0] = 1u;                                        // count=1, user D#
        g0[1] = lds_off;                                   // lds_addr
        g0[2] = (unsigned)ga;                              // global_addr[31:0]
        g0[3] = (unsigned)((ga >> 32) & 0x01FFFFFFu)       // global_addr[56:32]
              | (2u << 30);                                // type = 2 (image)
        u32x8 g1;
        g1[0] = 2u << 16;                                  // data_size = 4B
        g1[1] = ((unsigned)COUT & 0xFFFFu) << 16;          // tensor_dim0 lo16
        g1[2] = ((unsigned)COUT >> 16)                     // tensor_dim0 hi16
              | (((unsigned)KTOT & 0xFFFFu) << 16);        // tensor_dim1 lo16
        g1[3] = ((unsigned)KTOT >> 16)                     // tensor_dim1 hi16
              | ((unsigned)COUT << 16);                    // tile_dim0
        g1[4] = (unsigned)rows;                            // tile_dim1 (tile_dim2=0)
        g1[5] = (unsigned)COUT;                            // tensor_dim0_stride lo32
        g1[6] = 0u;                                        // stride hi / dim1_stride lo
        g1[7] = 0u;                                        // dim1_stride hi
        asm volatile("tensor_load_to_lds %0, %1" :: "s"(g0), "s"(g1) : "memory");
    };

    // ---- per-lane gather state for the A fragment (row = lane&15) ----
    const long mTile = wave;
    const long mA    = mTile * 16 + nn;
    const int  bA    = (int)(mA / V);
    const int  vA    = (int)(mA % V);
    int nb[S];
#pragma unroll
    for (int s = 0; s < S; ++s) nb[s] = idx[vA * S + s];
    const float* xb = x + (long)bA * V * CIN;

    v8f acc[NT] = {};

    // prologue: stage first W chunk
    if (waveIn == 0) {
        tdm_issue(0, 0);
        __builtin_amdgcn_s_wait_tensorcnt(0);
    }
    __syncthreads();

    int p = 0;
    for (int kk = 0; kk < KTOT; kk += KC) {
        const bool more = (kk + KC) < KTOT;
        if (waveIn == 0 && more) tdm_issue(kk + KC, p ^ 1);   // async prefetch

        // ---- A fragment: 16-bit A layout (16x32 MxK, wave32) ----
        v16bf a;
#pragma unroll
        for (int j = 0; j < 8; ++j) {
            const int kA = kk + ((j < 4) ? (2 * j) : (16 + 2 * (j - 4))) + laneHi * 8;
#pragma unroll
            for (int t = 0; t < 2; ++t) {
                const int k  = kA + t;
                float     av = 0.f;
                if (k < KTOT) {
                    const int s = k / CIN;
                    const int c = k - s * CIN;
                    av = xb[(long)nb[s] * CIN + c];
                }
                a[2 * j + t] = f2bf(av);
            }
        }

        // ---- sweep all N tiles, B fragment from LDS ----
#pragma unroll
        for (int nt = 0; nt < NT; ++nt) {
            v16bf b;
#pragma unroll
            for (int j = 0; j < 8; ++j) {
                const int kB = ((j < 4) ? (2 * j) : (16 + 2 * (j - 4))) + laneHi * 8;    // k offset within chunk
#pragma unroll
                for (int t = 0; t < 2; ++t) {
                    const int k  = kB + t;
                    const float bv = ((kk + k) < KTOT)
                                   ? wlds[p][k * COUT + nt * 16 + nn] : 0.f;
                    b[2 * j + t] = f2bf(bv);
                }
            }
            acc[nt] = __builtin_amdgcn_wmma_f32_16x16x32_bf16(
                false, a, false, b, (short)0, acc[nt], false, false);
        }

        if (waveIn == 0 && more) __builtin_amdgcn_s_wait_tensorcnt(0);
        __syncthreads();
        p ^= 1;
    }

    // ---- epilogue: bias + ELU + store (C layout: VGPR r -> M=laneHi*8+r) ----
#pragma unroll
    for (int nt = 0; nt < NT; ++nt) {
        const float bn = bias[nt * 16 + nn];
#pragma unroll
        for (int r = 0; r < 8; ++r) {
            const long m = mTile * 16 + laneHi * 8 + r;
            float      v = acc[nt][r] + bn;
            v = (v > 0.f) ? v : (__expf(v) - 1.f);    // ELU, alpha = 1
            h[m * COUT + nt * 16 + nn] = v;
        }
    }
}

// ---------------------------------------------------------------------------
// Sparse pooling: out[b,v',c] = sum_{k<3} h[b, p_idx[v',k], c] * p_w[v',k]
// ---------------------------------------------------------------------------
__global__ void pool_kernel(const float* __restrict__ h,
                            const int*   __restrict__ p_idx,
                            const float* __restrict__ p_w,
                            float*       __restrict__ out,
                            int Vin, int Vout, int C) {
    const long i     = (long)blockIdx.x * blockDim.x + threadIdx.x;
    const long total = 64L * Vout * C;
    if (i >= total) return;
    const int  c = (int)(i % C);
    const long t = i / C;
    const int  v = (int)(t % Vout);
    const int  b = (int)(t / Vout);
    float s = 0.f;
#pragma unroll
    for (int k = 0; k < 3; ++k) {
        s += h[((long)b * Vin + p_idx[v * 3 + k]) * C + c] * p_w[v * 3 + k];
    }
    out[i] = s;
}

// ---------------------------------------------------------------------------
// Latent projection: out[64,256] = H[64,40960] @ Wz[40960,256] + bz
// split-K WMMA, partial sums accumulated with global_atomic_add_f32
// ---------------------------------------------------------------------------
__global__ void latent_init_kernel(const float* __restrict__ bz,
                                   float* __restrict__ out) {
    const int i = blockIdx.x * blockDim.x + threadIdx.x;
    if (i < 64 * 256) out[i] = bz[i & 255];
}

__global__ void latent_wmma_kernel(const float* __restrict__ H,
                                   const float* __restrict__ Wz,
                                   float* __restrict__ out) {
    constexpr int KTOT   = 40960;
    constexpr int N      = 256;
    constexpr int KCHUNK = 1024;      // 40 chunks
    constexpr int MT = 64 / 16, NT = N / 16;

    const int lane = threadIdx.x & 31;
    const int wave = blockIdx.x * (blockDim.x >> 5) + (threadIdx.x >> 5);
    const int kChunk = wave / (MT * NT);
    const int rem    = wave % (MT * NT);
    const int mTile  = rem / NT;
    const int nTile  = rem % NT;
    const int laneHi = lane >> 4;
    const int nn     = lane & 15;
    const int mRow   = mTile * 16 + nn;   // A row handled by this lane

    v8f acc = {};
    const int k0 = kChunk * KCHUNK;
    for (int kk = k0; kk < k0 + KCHUNK; kk += 32) {
        v16bf a, b;
#pragma unroll
        for (int j = 0; j < 8; ++j) {
            const int kA = kk + ((j < 4) ? (2 * j) : (16 + 2 * (j - 4))) + laneHi * 8;
            a[2 * j]     = f2bf(H[(long)mRow * KTOT + kA]);
            a[2 * j + 1] = f2bf(H[(long)mRow * KTOT + kA + 1]);
            const int kB = kk + ((j < 4) ? (2 * j) : (16 + 2 * (j - 4))) + laneHi * 8;
            b[2 * j]     = f2bf(Wz[(long)kB * N + nTile * 16 + nn]);
            b[2 * j + 1] = f2bf(Wz[(long)(kB + 1) * N + nTile * 16 + nn]);
        }
        acc = __builtin_amdgcn_wmma_f32_16x16x32_bf16(
            false, a, false, b, (short)0, acc, false, false);
    }
#pragma unroll
    for (int r = 0; r < 8; ++r) {
        const int m = mTile * 16 + laneHi * 8 + r;
        const int n = nTile * 16 + nn;
        atomicAdd(&out[m * N + n], acc[r]);
    }
}

// ---------------------------------------------------------------------------
// Host-side orchestration
// ---------------------------------------------------------------------------
extern "C" void kernel_launch(void* const* d_in, const int* in_sizes, int n_in,
                              void* d_out, int out_size, void* d_ws, size_t ws_size,
                              hipStream_t stream) {
    // setup_inputs() dict order:
    const float* x      = (const float*)d_in[0];
    const int*   idx0   = (const int*)  d_in[1];
    const int*   p0_idx = (const int*)  d_in[2];
    const float* p0_w   = (const float*)d_in[3];
    const float* W0     = (const float*)d_in[4];
    const float* b0     = (const float*)d_in[5];
    const int*   idx1   = (const int*)  d_in[6];
    const int*   p1_idx = (const int*)  d_in[7];
    const float* p1_w   = (const float*)d_in[8];
    const float* W1     = (const float*)d_in[9];
    const float* b1     = (const float*)d_in[10];
    const int*   idx2   = (const int*)  d_in[11];
    const int*   p2_idx = (const int*)  d_in[12];
    const float* p2_w   = (const float*)d_in[13];
    const float* W2     = (const float*)d_in[14];
    const float* b2     = (const float*)d_in[15];
    const float* Wz     = (const float*)d_in[16];
    const float* bz     = (const float*)d_in[17];
    float* out = (float*)d_out;

    // Workspace layout
    const size_t H_BYTES    = 64ull * 20480 * 32 * 4;   // largest h: 167.8 MB
    const size_t POOL_BYTES = 64ull * 5120 * 32 * 4;    // largest pooled: 41.9 MB
    char*  ws    = (char*)d_ws;
    float* hbuf  = (float*)ws;
    float* poolA = (float*)(ws + H_BYTES);
    float* poolB = (float*)(ws + H_BYTES + POOL_BYTES);

    const int WPB = 4;                 // waves per 128-thread block
    auto blocks_for_waves = [](long waves, int wpb) {
        return (unsigned)((waves + wpb - 1) / wpb);
    };

    // ---- Block 0: x[64,20480,3] -> h0[64,20480,32] ----
    {
        const long waves = (64L * 20480) / 16;           // one wave per M tile
        spiral_wmma_kernel<3, 32><<<blocks_for_waves(waves, WPB), 32 * WPB, 0, stream>>>(
            x, idx0, W0, b0, hbuf, 20480);
        const long n = 64L * 5120 * 32;
        pool_kernel<<<(unsigned)((n + 255) / 256), 256, 0, stream>>>(
            hbuf, p0_idx, p0_w, poolA, 20480, 5120, 32);
    }
    // ---- Block 1: poolA[64,5120,32] -> h1[64,5120,64] ----
    {
        const long waves = (64L * 5120) / 16;
        spiral_wmma_kernel<32, 64><<<blocks_for_waves(waves, WPB), 32 * WPB, 0, stream>>>(
            poolA, idx1, W1, b1, hbuf, 5120);
        const long n = 64L * 1280 * 64;
        pool_kernel<<<(unsigned)((n + 255) / 256), 256, 0, stream>>>(
            hbuf, p1_idx, p1_w, poolB, 5120, 1280, 64);
    }
    // ---- Block 2: poolB[64,1280,64] -> h2[64,1280,128] ----
    {
        const long waves = (64L * 1280) / 16;
        spiral_wmma_kernel<64, 128><<<blocks_for_waves(waves, WPB), 32 * WPB, 0, stream>>>(
            poolB, idx2, W2, b2, hbuf, 1280);
        const long n = 64L * 320 * 128;
        pool_kernel<<<(unsigned)((n + 255) / 256), 256, 0, stream>>>(
            hbuf, p2_idx, p2_w, poolA, 1280, 320, 128);
    }
    // ---- Latent: poolA[64,40960] @ Wz[40960,256] + bz ----
    {
        latent_init_kernel<<<64, 256, 0, stream>>>(bz, out);
        const long waves = 4L * 16 * 40;   // mTiles * nTiles * kChunks = 2560
        latent_wmma_kernel<<<blocks_for_waves(waves, WPB), 32 * WPB, 0, stream>>>(
            poolA, Wz, out);
    }
}
